// InputScaledQuantLinear_1984274891103
// MI455X (gfx1250) — matrix operand order repaired and checked by reading, not compiled, not run on hardware
//
#include <hip/hip_runtime.h>
#include <hip/hip_bf16.h>

// ---------------------------------------------------------------------------
// y[n,o] = (dq_fp8(x[n,:]/s)*s) . W[o,:] + bias[o]
// bf16 WMMA GEMM. Block tile 128x256x64, 8 wave32 (2x4), wave tile 64x64.
// Double-buffered LDS pipeline (K-loop unrolled x2 for constant buffer
// addressing): B staged via GLOBAL_LOAD_ASYNC_TO_LDS_B128 (ASYNCcnt),
// A staged via VGPRs with fp8-e4m3 quant-dequant. One barrier per K-tile.
// ---------------------------------------------------------------------------

typedef __attribute__((ext_vector_type(16))) __bf16 v16bf;
typedef __attribute__((ext_vector_type(8)))  __bf16 v8bf;
typedef __attribute__((ext_vector_type(8)))  float  v8f;
typedef __attribute__((ext_vector_type(2)))  float  v2f;

#define BM 128
#define BN 256
#define BK 64
#define LDT 72   // padded LDS row stride (bf16): 144B = 36 dwords -> conflict-free walk

// ---- fp8 e4m3 quantize->dequantize --------------------------------------
#if defined(__has_builtin)
#if __has_builtin(__builtin_amdgcn_cvt_pk_fp8_f32) && __has_builtin(__builtin_amdgcn_cvt_pk_f32_fp8)
#define HW_FP8 1
#endif
#endif
#ifndef HW_FP8
#define HW_FP8 0
#endif

// ---- async global->LDS copy (CDNA5 ASYNCcnt path) ------------------------
#if defined(__has_builtin)
#if __has_builtin(__builtin_amdgcn_global_load_async_to_lds_b128)
#define HW_ASYNC 1
#endif
#endif
#ifndef HW_ASYNC
#define HW_ASYNC 0
#endif

// Compile-stderr probes (visible even on successful compile):
#if HW_FP8
#warning "CDNA5 probe: hardware fp8 cvt builtins ACTIVE"
#else
#warning "CDNA5 probe: fp8 cvt builtins NOT found -> software qdq path"
#endif
#if HW_ASYNC
#warning "CDNA5 probe: global_load_async_to_lds ACTIVE"
#else
#warning "CDNA5 probe: async-to-LDS builtin NOT found -> sync B staging"
#endif

__device__ __forceinline__ float qdq_sw(float v) {  // branchless RNE e4m3, sat 448
  unsigned int u = __float_as_uint(v);
  unsigned int a = u & 0x7FFFFFFFu;
  unsigned int t = (a + 0x0007FFFFu + ((a >> 20) & 1u)) & 0xFFF00000u;
  float rn = __uint_as_float(t);
  float rs = rintf(__uint_as_float(a) * 512.0f) * 0.001953125f;
  float r  = (a < 0x3C800000u) ? rs : rn;
  r = fminf(r, 448.0f);
  return copysignf(r, v);
}

__device__ __forceinline__ void qdq2(float f0, float f1, float& q0, float& q1) {
#if HW_FP8
  int p = __builtin_amdgcn_cvt_pk_fp8_f32(f0, f1, 0, false);  // v_cvt_pk_fp8_f32
  v2f r = __builtin_amdgcn_cvt_pk_f32_fp8(p, false);          // v_cvt_pk_f32_fp8
  q0 = r.x; q1 = r.y;
#else
  q0 = qdq_sw(f0);
  q1 = qdq_sw(f1);
#endif
}

#if HW_ASYNC
// Builtin signature (from hipcc diagnostic): pointer-to-int4, global (AS1) src
// and shared (AS3) dst, plus imm offset + imm cpol.
typedef int v4i __attribute__((vector_size(16)));
typedef __attribute__((address_space(1))) v4i g_v4i;
typedef __attribute__((address_space(3))) v4i l_v4i;
__device__ __forceinline__ void async_copy_b128(const void* g, void* l) {
  __builtin_amdgcn_global_load_async_to_lds_b128(
      (g_v4i*)(unsigned long long)g,
      (l_v4i*)(unsigned int)(unsigned long long)l, 0, 0);
}
__device__ __forceinline__ void wait_async_all() {
#if __has_builtin(__builtin_amdgcn_s_wait_asynccnt)
  __builtin_amdgcn_s_wait_asynccnt(0);
#else
  asm volatile("s_wait_asynccnt 0x0" ::: "memory");
#endif
}
#endif

__global__ __launch_bounds__(256)
void qlinear_wmma_bf16(const __hip_bfloat16* __restrict__ Xp,
                       const __hip_bfloat16* __restrict__ Wp,
                       const __hip_bfloat16* __restrict__ Bp,
                       const float* __restrict__ Sp,
                       __hip_bfloat16* __restrict__ Op,
                       int N, int K, int O) {
  const __bf16* X    = reinterpret_cast<const __bf16*>(Xp);
  const __bf16* W    = reinterpret_cast<const __bf16*>(Wp);
  const __bf16* bias = reinterpret_cast<const __bf16*>(Bp);
  __bf16*       out  = reinterpret_cast<__bf16*>(Op);

  __shared__ __bf16 lA[2][BM * LDT];   // 2 x 18 KB
  __shared__ __bf16 lB[2][BN * LDT];   // 2 x 36 KB

  const int tid  = threadIdx.x;
  const int lane = tid & 31;
  const int wave = tid >> 5;
  const int wm   = wave & 1;   // 2 waves over BM
  const int wn   = wave >> 1;  // 4 waves over BN
  const int lmod = lane & 15;
  const int lhi  = lane >> 4;

  const int m0 = blockIdx.y * BM;
  const int o0 = blockIdx.x * BN;

  const float s     = Sp[0];
  const float inv_s = 1.0f / s;

  const __bf16* Xb = X + (size_t)m0 * K;
  const __bf16* Wb = W + (size_t)o0 * K;

  // per-thread staging coordinates: chunk c -> (row = c>>3, col = (c&7)*8)
  const int arow[4] = { (tid + 0) >> 3, (tid + 256) >> 3, (tid + 512) >> 3, (tid + 768) >> 3 };
  const int acol    = (tid & 7) * 8;

  v8f acc[4][4];
  const v8f vzero = {0.f, 0.f, 0.f, 0.f, 0.f, 0.f, 0.f, 0.f};
#pragma unroll
  for (int i = 0; i < 4; ++i)
#pragma unroll
    for (int j = 0; j < 4; ++j) acc[i][j] = vzero;

  v8bf va[4];
#if !HW_ASYNC
  v8bf vb[8];
#endif

  // ---- pipeline stage helpers (buf is always a literal -> constant LDS addrs) ----
  auto stageB = [&](int kn, int buf) {
#pragma unroll
    for (int i = 0; i < 8; ++i) {
      const int c = tid + 256 * i;
#if HW_ASYNC
      async_copy_b128(Wb + (size_t)(c >> 3) * K + kn + (c & 7) * 8,
                      &lB[buf][(c >> 3) * LDT + (c & 7) * 8]);
#else
      vb[i] = *reinterpret_cast<const v8bf*>(Wb + (size_t)(c >> 3) * K + kn + (c & 7) * 8);
#endif
    }
  };
  auto loadA = [&](int kn) {
#pragma unroll
    for (int i = 0; i < 4; ++i)
      va[i] = *reinterpret_cast<const v8bf*>(Xb + (size_t)arow[i] * K + kn + acol);
  };
  auto storeStaged = [&](int buf) {
#pragma unroll
    for (int i = 0; i < 4; ++i) {  // quantize A, ds_store
      v8bf qa;
#pragma unroll
      for (int e = 0; e < 8; e += 2) {
        float q0, q1;
        qdq2((float)va[i][e] * inv_s, (float)va[i][e + 1] * inv_s, q0, q1);
        qa[e]     = (__bf16)(q0 * s);
        qa[e + 1] = (__bf16)(q1 * s);
      }
      *reinterpret_cast<v8bf*>(&lA[buf][arow[i] * LDT + acol]) = qa;
    }
#if !HW_ASYNC
#pragma unroll
    for (int i = 0; i < 8; ++i) {
      const int c = tid + 256 * i;
      *reinterpret_cast<v8bf*>(&lB[buf][(c >> 3) * LDT + (c & 7) * 8]) = vb[i];
    }
#endif
  };
  auto compute = [&](int buf) {
#pragma unroll
    for (int ks = 0; ks < BK; ks += 32) {
      v16bf afrag[4];
#pragma unroll
      for (int i = 0; i < 4; ++i) {  // A: lane<16 holds K[ks..ks+7],K[ks+16..23]
        const int m    = wm * 64 + i * 16 + lmod;
        const int kofs = ks + lhi * 8;
        v8bf c0 = *reinterpret_cast<const v8bf*>(&lA[buf][m * LDT + kofs]);
        v8bf c1 = *reinterpret_cast<const v8bf*>(&lA[buf][m * LDT + kofs + 16]);
        afrag[i] = __builtin_shufflevector(c0, c1, 0, 1, 2, 3, 4, 5, 6, 7,
                                           8, 9, 10, 11, 12, 13, 14, 15);
      }
#pragma unroll
      for (int j = 0; j < 4; ++j) {  // B: lane<16 holds K[ks..ks+7],K[ks+16..23] of one column
        const int o    = wn * 64 + j * 16 + lmod;
        const int kofs = ks + lhi * 8;
        v8bf c0 = *reinterpret_cast<const v8bf*>(&lB[buf][o * LDT + kofs]);
        v8bf c1 = *reinterpret_cast<const v8bf*>(&lB[buf][o * LDT + kofs + 16]);
        v16bf bfrag = __builtin_shufflevector(c0, c1, 0, 1, 2, 3, 4, 5, 6, 7,
                                              8, 9, 10, 11, 12, 13, 14, 15);
#pragma unroll
        for (int i = 0; i < 4; ++i)
          acc[i][j] = __builtin_amdgcn_wmma_f32_16x16x32_bf16(
              false, afrag[i], false, bfrag, (short)0, acc[i][j], false, false);
      }
    }
  };
  auto finish_stage = [&]() {
#if HW_ASYNC
    wait_async_all();
#endif
    __syncthreads();
  };

  // ---- prologue: stage K-tile 0 into buffer 0 ----
  stageB(0, 0);
  loadA(0);
  storeStaged(0);
  finish_stage();

  // ---- main pipeline, unrolled x2 so buffer indices are literals ----
  for (int k0 = 0; k0 < K; k0 += 2 * BK) {
    {  // phase 0: compute buf0, stage k0+BK -> buf1
      const int kn = k0 + BK;
      const bool hn = kn < K;
      if (hn) { stageB(kn, 1); loadA(kn);
                __builtin_prefetch(Xb + (size_t)arow[0] * K + kn + BK + acol, 0, 0); }
      compute(0);
      if (hn) storeStaged(1);
      finish_stage();
    }
    if (k0 + BK < K) {  // phase 1: compute buf1, stage k0+2BK -> buf0
      const int kn = k0 + 2 * BK;
      const bool hn = kn < K;
      if (hn) { stageB(kn, 0); loadA(kn);
                __builtin_prefetch(Xb + (size_t)arow[0] * K + kn + BK + acol, 0, 0); }
      compute(1);
      if (hn) storeStaged(0);
      finish_stage();
    }
  }

  // ---- epilogue: acc + bias -> bf16, non-temporal stores ----
#pragma unroll
  for (int j = 0; j < 4; ++j) {
    const int o    = o0 + wn * 64 + j * 16 + lmod;
    const float bj = (float)bias[o];
#pragma unroll
    for (int i = 0; i < 4; ++i) {
      const int mbase = m0 + wm * 64 + i * 16 + lhi * 8;
#pragma unroll
      for (int r = 0; r < 8; ++r) {
        __builtin_nontemporal_store((__bf16)(acc[i][j][r] + bj),
                                    &out[(size_t)(mbase + r) * O + o]);
      }
    }
  }
}

extern "C" void kernel_launch(void* const* d_in, const int* in_sizes, int n_in,
                              void* d_out, int out_size, void* d_ws, size_t ws_size,
                              hipStream_t stream) {
  const __hip_bfloat16* X = (const __hip_bfloat16*)d_in[0];
  const __hip_bfloat16* W = (const __hip_bfloat16*)d_in[1];
  const __hip_bfloat16* B = (const __hip_bfloat16*)d_in[2];
  const float*          S = (const float*)d_in[3];
  __hip_bfloat16*     out = (__hip_bfloat16*)d_out;

  const int O = in_sizes[2];              // 2048
  const int K = in_sizes[1] / O;          // 2048
  const int N = in_sizes[0] / K;          // 32768

  dim3 grid(O / BN, N / BM);              // 8 x 256 blocks
  qlinear_wmma_bf16<<<grid, 256, 0, stream>>>(X, W, B, S, out, N, K, O);
}
